// DockBoardAttention_38689065402575
// MI455X (gfx1250) — matrix operand order repaired and checked by reading, not compiled, not run on hardware
//
#include <hip/hip_runtime.h>
#include <cstdint>
#include <cstddef>

// MI455X / gfx1250, wave32. One batch per wave.
// HBM floor ~24us (570MB @ 23.3TB/s); K/V projections on v_wmma_f32_16x16x32_f16,
// grid_feat staged via TENSOR_LOAD_TO_LDS (TDM) with LDS row padding (stride 65 dwords),
// attention tail on VALU + shfl_xor reductions.

typedef __attribute__((ext_vector_type(16))) _Float16 v16h;
typedef __attribute__((ext_vector_type(8)))  float    v8f;
typedef __attribute__((ext_vector_type(4)))  uint32_t u32x4;
typedef __attribute__((ext_vector_type(8)))  int      i32x8;
typedef __attribute__((ext_vector_type(4)))  int      i32x4;

#define WAVES_PER_BLOCK 4

__global__ __launch_bounds__(WAVES_PER_BLOCK * 32)
void dock_attn_kernel(const float* __restrict__ dock, const float* __restrict__ grid,
                      const float* __restrict__ Wq, const float* __restrict__ bq,
                      const float* __restrict__ Wk, const float* __restrict__ bk,
                      const float* __restrict__ Wv, const float* __restrict__ bv,
                      const float* __restrict__ Wo, const float* __restrict__ bo,
                      float* __restrict__ out, int nbatch)
{
    // Per-wave private LDS regions (no cross-wave sharing -> no block barriers).
    __shared__ float gbuf[WAVES_PER_BLOCK][32 * 65]; // 64 dwords/row + 1 pad dword (TDM pad)
    __shared__ float dkb [WAVES_PER_BLOCK][80];      // dock_masks[b] (75 f32)
    __shared__ float qbf [WAVES_PER_BLOCK][64];      // Q broadcast (48 f32)

    const int tid  = threadIdx.x;
    const int w    = tid >> 5;
    const int lane = tid & 31;
    const int lo   = lane & 15;   // column / row-in-tile index
    const int hi   = lane >> 4;   // half-wave select

    // ---------------- one-time weight preload ----------------
    // A operands (16x32 f16, ISA A-layout): lane holds row m=lo;
    // hi=0 -> K {0..7,16..23}, hi=1 -> K {8..15,24..31}.
    v16h ka, va;
    {
        const float* kr = Wk + lo * 32 + 8 * hi;
        const float* vr = Wv + lo * 32 + 8 * hi;
#pragma unroll
        for (int i = 0; i < 8; ++i) {
            ka[i]     = (_Float16)kr[i];
            ka[i + 8] = (_Float16)kr[16 + i];
            va[i]     = (_Float16)vr[i];
            va[i + 8] = (_Float16)vr[16 + i];
        }
    }
    // Bias folded into WMMA C operand: C[r] corresponds to row m = r + 8*hi.
    v8f ck, cv;
#pragma unroll
    for (int r = 0; r < 8; ++r) { ck[r] = bk[8 * hi + r]; cv[r] = bv[8 * hi + r]; }

    float wq[25];                         // Wq row d = lo
#pragma unroll
    for (int c = 0; c < 25; ++c) wq[c] = Wq[lo * 25 + c];
    const float bq_s = bq[lo];

    float wo[8];                          // Wo[lo][8*hi + r]
#pragma unroll
    for (int r = 0; r < 8; ++r) wo[r] = Wo[lo * 16 + 8 * hi + r];
    const float bo_s = bo[lo];

    const uint32_t lds_off = (uint32_t)(uintptr_t)(&gbuf[w][0]); // addr[31:0] == LDS offset

    const int gwave  = blockIdx.x * WAVES_PER_BLOCK + w;
    const int nwaves = gridDim.x * WAVES_PER_BLOCK;

    for (int b = gwave; b < nbatch; b += nwaves) {
        // ---- TDM: grid_feat[b] (32x64 f32, row-major) -> LDS, 1-dword pad per 64 ----
        {
            const uint64_t ga = (uint64_t)(uintptr_t)(grid + (size_t)b * 2048);
            u32x4 g0;
            g0[0] = 1u;                                            // count=1, is_restore=0, no gather
            g0[1] = lds_off;                                       // lds_addr
            g0[2] = (uint32_t)(ga & 0xffffffffu);                  // global_addr[31:0]
            g0[3] = (uint32_t)((ga >> 32) & 0x1ffffffu) | (2u << 30); // addr[56:32] | type=2
            i32x8 g1;
            g1[0] = (int)((2u << 16) | (1u << 20) | (5u << 22));   // 4B elems, pad_en, every 64dw, pad 1dw
            g1[1] = (int)(64u << 16);                              // tensor_dim0 = 64
            g1[2] = (int)(32u << 16);                              // tensor_dim1 = 32
            g1[3] = (int)(64u << 16);                              // tile_dim0 = 64
            g1[4] = 32;                                            // tile_dim1 = 32
            g1[5] = 64;                                            // tensor_dim0_stride = 64
            g1[6] = 0;
            g1[7] = 0;
            i32x4 gz = {0, 0, 0, 0};
#if __has_include(<hip/amd_detail/amd_gfx1250_TDM.h>)
            i32x8 gz8 = {0, 0, 0, 0, 0, 0, 0, 0};
            __builtin_amdgcn_tensor_load_to_lds(g0, g1, gz, gz, gz8, 0);
#else
            __builtin_amdgcn_tensor_load_to_lds(g0, g1, gz, gz, 0);
#endif
        }

        // ---- stage dock_masks[b] (75 f32) coalesced into LDS ----
#pragma unroll
        for (int j = 0; j < 3; ++j) {
            int i = j * 32 + lane;
            if (i < 75) dkb[w][i] = dock[(size_t)b * 75 + i];
        }

        // ---- Q projection (d = lo): row q=hi and row q=2 per lane ----
        float qa = bq_s, q2 = bq_s;
#pragma unroll
        for (int c = 0; c < 25; ++c) {
            const float t = wq[c];
            qa = fmaf(dkb[w][hi * 25 + c], t, qa);
            q2 = fmaf(dkb[w][50 + c],      t, q2);
        }
        qbf[w][hi * 16 + lo] = qa;   // Q[0][lo] / Q[1][lo]
        qbf[w][32 + lo]      = q2;   // Q[2][lo] (duplicate same-value writes benign)

        // transpose read: lane needs Q[q][8*hi + j]
        float qn[3][8];
#pragma unroll
        for (int q = 0; q < 3; ++q)
#pragma unroll
            for (int j = 0; j < 8; ++j)
                qn[q][j] = qbf[w][q * 16 + 8 * hi + j];

        // next-iteration dock prefetch (speculative, dropped if OOB)
        if (b + nwaves < nbatch)
            __builtin_prefetch(dock + (size_t)(b + nwaves) * 75, 0, 1);

        // ---- wait for TDM, then K/V projections: 4 N-tiles x {K,V} WMMAs ----
        __builtin_amdgcn_s_wait_tensorcnt(0);
        asm volatile("" ::: "memory");

        v8f kt[4], vt[4];
#pragma unroll
        for (int t = 0; t < 4; ++t) {
            // B operand (32x16 f16): lane column n = 16t+lo, K = 8*hi + (i&7) + 16*(i>>3),
            // padded LDS stride 65 dwords -> conflict-free banks.
            v16h bmat;
#pragma unroll
            for (int i = 0; i < 16; ++i)
                bmat[i] = (_Float16)gbuf[w][(hi * 8 + (i & 7) + ((i >> 3) << 4)) * 65 + t * 16 + lo];
            kt[t] = __builtin_amdgcn_wmma_f32_16x16x32_f16(false, ka, false, bmat,
                                                           (short)0, ck, false, false);
            vt[t] = __builtin_amdgcn_wmma_f32_16x16x32_f16(false, va, false, bmat,
                                                           (short)0, cv, false, false);
        }

        // ---- scores S[q][n] (n = 16t+lo), softmax over 64 columns ----
        float attn[3][4];
#pragma unroll
        for (int q = 0; q < 3; ++q) {
            float p[4];
#pragma unroll
            for (int t = 0; t < 4; ++t) {
                float acc = 0.f;
#pragma unroll
                for (int r = 0; r < 8; ++r) acc = fmaf(qn[q][r], kt[t][r], acc);
                acc += __shfl_xor(acc, 16, 32);     // join m-halves (same n in lane pair)
                p[t] = acc * 0.25f;                 // 1/sqrt(HD)
            }
            float m = fmaxf(fmaxf(p[0], p[1]), fmaxf(p[2], p[3]));
#pragma unroll
            for (int msk = 1; msk < 16; msk <<= 1) m = fmaxf(m, __shfl_xor(m, msk, 32));
            float s = 0.f;
#pragma unroll
            for (int t = 0; t < 4; ++t) { p[t] = __expf(p[t] - m); s += p[t]; }
#pragma unroll
            for (int msk = 1; msk < 16; msk <<= 1) s += __shfl_xor(s, msk, 32);
            const float inv = 1.0f / s;
#pragma unroll
            for (int t = 0; t < 4; ++t) attn[q][t] = p[t] * inv;
        }

        // ---- ctx[q][m] = sum_n attn*V, then output projection ----
        float outv[3];
#pragma unroll
        for (int q = 0; q < 3; ++q) {
            float po = 0.f;
#pragma unroll
            for (int r = 0; r < 8; ++r) {                 // m = 8*hi + r
                float pc = 0.f;
#pragma unroll
                for (int t = 0; t < 4; ++t) pc = fmaf(attn[q][t], vt[t][r], pc);
#pragma unroll
                for (int msk = 1; msk < 16; msk <<= 1) pc += __shfl_xor(pc, msk, 32);
                po = fmaf(pc, wo[r], po);                 // partial over this m-half
            }
            po += __shfl_xor(po, 16, 32);                 // join m-halves (e = lo in both)
            outv[q] = po + bo_s;
        }

        // ---- store out[b][q*16 + e], e = lo ----
        float* ob = out + (size_t)b * 48;
        ob[hi * 16 + lo] = hi ? outv[1] : outv[0];
        if (hi == 0) ob[32 + lo] = outv[2];
    }
}

extern "C" void kernel_launch(void* const* d_in, const int* in_sizes, int n_in,
                              void* d_out, int out_size, void* d_ws, size_t ws_size,
                              hipStream_t stream) {
    const float* dock = (const float*)d_in[0];
    const float* grid = (const float*)d_in[1];
    const float* Wq   = (const float*)d_in[2];
    const float* bq   = (const float*)d_in[3];
    const float* Wk   = (const float*)d_in[4];
    const float* bk   = (const float*)d_in[5];
    const float* Wv   = (const float*)d_in[6];
    const float* bv   = (const float*)d_in[7];
    const float* Wo   = (const float*)d_in[8];
    const float* bo   = (const float*)d_in[9];
    float* out = (float*)d_out;

    const int nbatch = in_sizes[0] / 75;   // B*3*25
    int blocks = 2048;                     // 8192 waves -> 8 batches/wave at B=65536
    if (blocks * WAVES_PER_BLOCK > nbatch)
        blocks = (nbatch + WAVES_PER_BLOCK - 1) / WAVES_PER_BLOCK;

    dock_attn_kernel<<<blocks, WAVES_PER_BLOCK * 32, 0, stream>>>(
        dock, grid, Wq, bq, Wk, bk, Wv, bv, Wo, bo, out, nbatch);
}
